// Graph_LODE_21019569947018
// MI455X (gfx1250) — compile-verified, matrix operand order checked by reading, not run on hardware
//
#include <hip/hip_runtime.h>
#include <hip/hip_bf16.h>
#include <math.h>

// ---------------- model constants ----------------
#define B_   4
#define T_   100
#define L_   64

typedef __attribute__((ext_vector_type(16))) __bf16 v16bf;
typedef __attribute__((ext_vector_type(8)))  float  v8f;

// =================================================================
// Templated WMMA GEMM:  C[M,N] = act( A[M,K] * B[K,N] + bias[N] )
// K, N compile-time => B-load strides become immediate offsets and
// all K-tail guards fold at compile time. A,B,C row-major fp32;
// bf16 fragments, fp32 accumulation via v_wmma_f32_16x16x32_bf16.
// One wave per 16x16 tile, batch via blockIdx.z.
// act: 0=none, 1=elu, 2=tanh
// =================================================================
template<int K, int N>
__global__ void __launch_bounds__(32)
wmma_gemm_kernel(const float* __restrict__ A, const float* __restrict__ Bm,
                 const float* __restrict__ bias, float* __restrict__ C,
                 int M, long long sA, long long sB, long long sC, int act)
{
    const int lane = threadIdx.x;              // 0..31, full wave32
    const int tm = blockIdx.x, tn = blockIdx.y, bz = blockIdx.z;
    A  += sA * (long long)bz;
    Bm += sB * (long long)bz;
    C  += sC * (long long)bz;

    const int half  = lane >> 4;               // 0 | 1
    const int l15   = lane & 15;
    const int baseK = half * 8;                // K sub-block per half-wave
    const int arow  = tm * 16 + l15;           // A row held by this lane
    const int bcol  = tn * 16 + l15;           // B col held by this lane

    v8f acc = {0.f, 0.f, 0.f, 0.f, 0.f, 0.f, 0.f, 0.f};

    // wave-uniform fast/slow selection (N/K parts fold at compile time)
    const bool interior = ((tm * 16 + 16) <= M) && ((tn * 16 + 16) <= N) &&
                          ((K & 31) == 0);

    if (interior) {
        const float* __restrict__ ap = A + (long long)arow * K + baseK;
        const float* __restrict__ bp = Bm + (long long)baseK * N + bcol;
        for (int k0 = 0; k0 < K; k0 += 32) {
            // A fragment: elements 0..7 = K baseK+0..7 ; 8..15 = K baseK+16..23
            const float4 a0 = *(const float4*)(ap);
            const float4 a1 = *(const float4*)(ap + 4);
            const float4 a2 = *(const float4*)(ap + 16);
            const float4 a3 = *(const float4*)(ap + 20);
            float bvv[16];
#pragma unroll
            for (int e = 0; e < 16; ++e) {
                const int krel = (e & 7) + ((e >> 3) << 4);   // constant
                bvv[e] = bp[krel * N];                        // immediate offset
            }
            ap += 32;
            bp += 32 * N;
            v16bf af, bf;
            af[0] = (__bf16)a0.x;  af[1] = (__bf16)a0.y;
            af[2] = (__bf16)a0.z;  af[3] = (__bf16)a0.w;
            af[4] = (__bf16)a1.x;  af[5] = (__bf16)a1.y;
            af[6] = (__bf16)a1.z;  af[7] = (__bf16)a1.w;
            af[8] = (__bf16)a2.x;  af[9] = (__bf16)a2.y;
            af[10] = (__bf16)a2.z; af[11] = (__bf16)a2.w;
            af[12] = (__bf16)a3.x; af[13] = (__bf16)a3.y;
            af[14] = (__bf16)a3.z; af[15] = (__bf16)a3.w;
#pragma unroll
            for (int e = 0; e < 16; ++e) bf[e] = (__bf16)bvv[e];
            acc = __builtin_amdgcn_wmma_f32_16x16x32_bf16(
                      false, af, false, bf, (short)0, acc, false, false);
        }
    } else {
        // edge tiles / tiny K: clamp row/col once, K-guard folds statically
        const int   arc = (arow < M) ? arow : (M - 1);
        const float amk = (arow < M) ? 1.f : 0.f;
        const int   bcc = (bcol < N) ? bcol : (N - 1);
        const float bmk = (bcol < N) ? 1.f : 0.f;
        for (int k0 = 0; k0 < K; k0 += 32) {
            v16bf af, bf;
#pragma unroll
            for (int e = 0; e < 16; ++e) {
                const int kk = k0 + baseK + (e & 7) + ((e >> 3) << 4);
                float av = 0.f, bv = 0.f;
                if (kk < K) {                   // compile-time foldable
                    av = A[(long long)arc * K + kk] * amk;
                    bv = Bm[(long long)kk * N + bcc] * bmk;
                }
                af[e] = (__bf16)av;
                bf[e] = (__bf16)bv;
            }
            acc = __builtin_amdgcn_wmma_f32_16x16x32_bf16(
                      false, af, false, bf, (short)0, acc, false, false);
        }
    }

    if (bcol < N) {
        const float badd = bias ? bias[bcol] : 0.f;
#pragma unroll
        for (int r = 0; r < 8; ++r) {
            const int row = tm * 16 + half * 8 + r;   // C/D VGPR layout
            if (row < M) {
                float v = acc[r] + badd;
                if      (act == 1) v = (v > 0.f) ? v : (expf(v) - 1.f);
                else if (act == 2) v = tanhf(v);
                C[(long long)row * N + bcol] = v;
            }
        }
    }
}

// =================================================================
// Edge scatter:  base[b,dst,t,c] += xm[b,src,t,c] * ea[e]
// =================================================================
__global__ void scatter_edges_kernel(const float* __restrict__ xm,
                                     const int* __restrict__ ei,
                                     const float* __restrict__ ea,
                                     float* __restrict__ base,
                                     int E, int Nn, int C)
{
    long long i = (long long)blockIdx.x * blockDim.x + threadIdx.x;
    const long long total = (long long)E * B_ * T_ * C;
    if (i >= total) return;
    const int c = (int)(i % C);
    const int t = (int)((i / C) % T_);
    const int e = (int)((i / ((long long)C * T_)) % E);
    const int b = (int)(i / ((long long)C * T_ * E));
    const int src = ei[e], dst = ei[E + e];
    const float v = xm[(((long long)b * Nn + src) * T_ + t) * C + c] * ea[e];
    unsafeAtomicAdd(&base[(((long long)b * Nn + dst) * T_ + t) * C + c], v);
}

__global__ void elu_kernel(float* __restrict__ x, long long n)
{
    long long i = (long long)blockIdx.x * blockDim.x + threadIdx.x;
    if (i < n) { float v = x[i]; x[i] = (v > 0.f) ? v : (expf(v) - 1.f); }
}

__global__ void zero_kernel(float* __restrict__ x, long long n)
{
    long long i = (long long)blockIdx.x * blockDim.x + threadIdx.x;
    if (i < n) x[i] = 0.f;
}

// z = mu + eps * exp(0.5*logvar)
__global__ void reparam_kernel(const float* __restrict__ mu, const float* __restrict__ lv,
                               const float* __restrict__ eps, float* __restrict__ z,
                               long long n)
{
    long long i = (long long)blockIdx.x * blockDim.x + threadIdx.x;
    if (i < n) z[i] = mu[i] + eps[i] * expf(0.5f * lv[i]);
}

// hinv scatter: xb = [zeros(B,128,L); z(B,128,L)]; agg[b,dst,c] += xb[b,src,c]*ea[e]
__global__ void hinv_scatter_kernel(const float* __restrict__ z,
                                    const int* __restrict__ ei,
                                    const float* __restrict__ ea,
                                    float* __restrict__ agg, int E)
{
    long long i = (long long)blockIdx.x * blockDim.x + threadIdx.x;
    const long long total = (long long)E * B_ * L_;
    if (i >= total) return;
    const int c = (int)(i % L_);
    const int e = (int)((i / L_) % E);
    const int b = (int)(i / ((long long)L_ * E));
    const int src = ei[e], dst = ei[E + e];
    if (src >= 128) {
        const float v = z[((long long)b * 128 + (src - 128)) * L_ + c] * ea[e];
        unsafeAtomicAdd(&agg[((long long)b * 256 + dst) * L_ + c], v);
    }
}

// =================================================================
// Reverse GRU over time; one block (64 threads) per (b,n) row.
// X layout [B*N, T, L]; rows are independent across time.
// =================================================================
__global__ void __launch_bounds__(64)
rev_gru_kernel(const float* __restrict__ X, const float* __restrict__ W,
               const float* __restrict__ U, const float* __restrict__ bg,
               float* __restrict__ H)
{
    const int row = blockIdx.x;
    const int j   = threadIdx.x;
    __shared__ float h[L_], xt[L_];
    h[j] = 0.f;
    __syncthreads();
    for (int t = T_ - 1; t >= 0; --t) {
        xt[j] = X[((long long)row * T_ + t) * L_ + j];
        __syncthreads();
        float az = 0.f, ar = 0.f, an = 0.f, uz = 0.f, ur = 0.f, un = 0.f;
#pragma unroll 4
        for (int k = 0; k < L_; ++k) {
            const float xk = xt[k], hk = h[k];
            az += xk * W[          k * L_ + j];
            ar += xk * W[4096 +    k * L_ + j];
            an += xk * W[8192 +    k * L_ + j];
            uz += hk * U[          k * L_ + j];
            ur += hk * U[4096 +    k * L_ + j];
            un += hk * U[8192 +    k * L_ + j];
        }
        const float zg = 1.f / (1.f + expf(-(az + uz + bg[j])));
        const float rg = 1.f / (1.f + expf(-(ar + ur + bg[L_ + j])));
        const float ng = tanhf(an + rg * un + bg[2 * L_ + j]);
        const float nh = (1.f - zg) * ng + zg * h[j];
        __syncthreads();
        h[j] = nh;
        __syncthreads();
    }
    H[(long long)row * L_ + j] = h[j];
}

// =================================================================
// Neural ODE (RK4) trajectory; one block (128 threads) per (b,n) row.
// f(z) = tanh(z@W1 + b1)@W2 + b2,  W1:[64,128], W2:[128,64].
// traj layout [B*128, T, 64].
// =================================================================
__device__ __forceinline__ void gde_eval(const float* __restrict__ W1,
                                         const float* __restrict__ b1,
                                         const float* __restrict__ W2,
                                         const float* __restrict__ b2,
                                         const float* v, float* hid, float* out, int j)
{
    float a = b1[j];
#pragma unroll 8
    for (int k = 0; k < 64; ++k) a += v[k] * W1[k * 128 + j];
    hid[j] = tanhf(a);
    __syncthreads();
    if (j < 64) {
        float o = b2[j];
#pragma unroll 8
        for (int k = 0; k < 128; ++k) o += hid[k] * W2[k * 64 + j];
        out[j] = o;
    }
    __syncthreads();
}

__global__ void __launch_bounds__(128)
gde_kernel(const float* __restrict__ xb2, const float* __restrict__ W1,
           const float* __restrict__ b1, const float* __restrict__ W2,
           const float* __restrict__ b2, float* __restrict__ traj)
{
    const int row = blockIdx.x;              // b*128 + n
    const int b = row >> 7, n = row & 127;
    const int j = threadIdx.x;
    __shared__ float z[64], v[64], hid[128], k1[64], k2[64], k3[64], k4[64];
    if (j < 64) {
        z[j] = xb2[((long long)b * 256 + n) * 64 + j];
        traj[(long long)row * T_ * 64 + j] = z[j];
    }
    __syncthreads();
    const float dt = 1.f / (float)(T_ - 1);
    for (int s = 1; s < T_; ++s) {
        gde_eval(W1, b1, W2, b2, z, hid, k1, j);
        if (j < 64) v[j] = z[j] + 0.5f * dt * k1[j];
        __syncthreads();
        gde_eval(W1, b1, W2, b2, v, hid, k2, j);
        if (j < 64) v[j] = z[j] + 0.5f * dt * k2[j];
        __syncthreads();
        gde_eval(W1, b1, W2, b2, v, hid, k3, j);
        if (j < 64) v[j] = z[j] + dt * k3[j];
        __syncthreads();
        gde_eval(W1, b1, W2, b2, v, hid, k4, j);
        if (j < 64) {
            z[j] += (dt / 6.f) * (k1[j] + 2.f * k2[j] + 2.f * k3[j] + k4[j]);
            traj[((long long)row * T_ + s) * 64 + j] = z[j];
        }
        __syncthreads();
    }
}

// =================================================================
// host-side helpers
// =================================================================
template<int K, int N>
static inline void launch_gemm(hipStream_t s, const float* A, const float* Bm,
                               const float* bias, float* C, int M,
                               int batch, long long sA, long long sB, long long sC,
                               int act)
{
    dim3 g((unsigned)((M + 15) / 16), (unsigned)((N + 15) / 16), (unsigned)batch);
    wmma_gemm_kernel<K, N><<<g, dim3(32), 0, s>>>(A, Bm, bias, C, M, sA, sB, sC, act);
}

// graph conv: base = x@Wr + b; xm = x@Wm; base += scatter(xm); elu(base)
template<int Cin, int Cout>
static inline void gconv(hipStream_t s, const float* x, const int* ei, const float* ea,
                         int E, const float* Wr, const float* Wm, const float* bias,
                         int Nn, float* xm, float* base)
{
    const int Mrows = B_ * Nn * T_;
    launch_gemm<Cin, Cout>(s, x, Wm, nullptr, xm,   Mrows, 1, 0, 0, 0, 0);
    launch_gemm<Cin, Cout>(s, x, Wr, bias,    base, Mrows, 1, 0, 0, 0, 0);
    const long long te = (long long)E * B_ * T_ * Cout;
    scatter_edges_kernel<<<(unsigned)((te + 255) / 256), 256, 0, s>>>(xm, ei, ea, base, E, Nn, Cout);
    const long long n = (long long)Mrows * Cout;
    elu_kernel<<<(unsigned)((n + 255) / 256), 256, 0, s>>>(base, n);
}

extern "C" void kernel_launch(void* const* d_in, const int* in_sizes, int n_in,
                              void* d_out, int out_size, void* d_ws, size_t ws_size,
                              hipStream_t stream)
{
    (void)in_sizes; (void)n_in; (void)out_size; (void)ws_size;
    auto F = [&](int i) { return (const float*)d_in[i]; };
    auto I = [&](int i) { return (const int*)d_in[i]; };

    // ---- inputs (setup_inputs() dict order) ----
    const float* phi_t = F(0);   const float* eps = F(1);
    const int* tg_ei  = I(2);  const float* tg_ea  = F(3);
    const int* tg1_ei = I(4);  const float* tg1_ea = F(5);
    const int* tg2_ei = I(6);  const float* tg2_ea = F(7);
    const int* bg_ei  = I(8);  const float* bg_ea  = F(9);
    const int* bg1_ei = I(10); const float* bg1_ea = F(11);
    const int* bg2_ei = I(12); const float* bg2_ea = F(13);
    const int* bg3_ei = I(14); const float* bg3_ea = F(15);
    const int* hinv_ei = I(16); const float* hinv_ea = F(17);
    const float* t_P01 = F(18); const float* t_P12 = F(19); const float* t_P23 = F(20);
    const float* P43 = F(21); const float* P32 = F(22); const float* P21 = F(23); const float* P10 = F(24);
    const float* h_L = F(25); const float* H_mat = F(26);
    const float* c1_Wr = F(27); const float* c1_Wm = F(28); const float* c1_b = F(29);
    const float* c2_Wr = F(30); const float* c2_Wm = F(31); const float* c2_b = F(32);
    const float* c3_Wr = F(33); const float* c3_Wm = F(34); const float* c3_b = F(35);
    const float* d4_Wr = F(36); const float* d4_Wm = F(37); const float* d4_b = F(38);
    const float* d3_Wr = F(39); const float* d3_Wm = F(40); const float* d3_b = F(41);
    const float* d2_Wr = F(42); const float* d2_Wm = F(43); const float* d2_b = F(44);
    const float* d1_Wr = F(45); const float* d1_Wm = F(46); const float* d1_b = F(47);
    const float* fce1_W = F(48); const float* fce1_b = F(49);
    const float* fce2_W = F(50); const float* fce2_b = F(51);
    const float* gru_W = F(52); const float* gru_U = F(53); const float* gru_b = F(54);
    const float* mu_W = F(55); const float* mu_b = F(56);
    const float* lv_W = F(57); const float* lv_b = F(58);
    const float* trans_W = F(59);
    const float* gde_W1 = F(60); const float* gde_b1 = F(61);
    const float* gde_W2 = F(62); const float* gde_b2 = F(63);
    const float* fcd3_W = F(64); const float* fcd3_b = F(65);
    const float* fcd4_W = F(66); const float* fcd4_b = F(67);

    // ---- outputs (phi_h, phi_t_, l_h, mu, logvar) ----
    float* out       = (float*)d_out;
    float* out_phi_h = out;                 // B*2048*100 = 819200
    float* out_phi_t = out + 819200;        // B*1024*100 = 409600
    float* out_l_h   = out + 1228800;       // 819200
    float* out_mu    = out + 2048000;       // 32768
    float* out_lv    = out + 2080768;       // 32768

    // ---- workspace layout (floats) ----
    float* ws   = (float*)d_ws;
    float* buf0 = ws;                       // 6,553,600
    float* buf1 = ws + 6553600;             // 6,553,600
    float* buf2 = ws + 13107200;            // 6,553,600
    float* hgru = ws + 19660800;            // 32768
    float* zlat = hgru + 32768;             // 32768
    float* agg  = zlat + 32768;             // 65536
    float* xb2  = agg  + 65536;             // 65536
    float* traj = xb2  + 65536;             // 3,276,800

    hipStream_t s = stream;

    // =================== encoder ===================
    // gconv1 (N=1024, 1->16) ; layout [B,N,T,C]
    gconv<1, 16>(s, phi_t, tg_ei, tg_ea, 8192, c1_Wr, c1_Wm, c1_b, 1024, buf0, buf1);
    // pool t_P01: [512,1024] x [1024,1600] per batch
    launch_gemm<1024, 1600>(s, t_P01, buf1, nullptr, buf2, 512, B_, 0, 1638400, 819200, 0);
    // gconv2 (N=512, 16->32)
    gconv<16, 32>(s, buf2, tg1_ei, tg1_ea, 4096, c2_Wr, c2_Wm, c2_b, 512, buf0, buf1);
    launch_gemm<512, 3200>(s, t_P12, buf1, nullptr, buf2, 256, B_, 0, 1638400, 819200, 0);
    // gconv3 (N=256, 32->64)
    gconv<32, 64>(s, buf2, tg2_ei, tg2_ea, 2048, c3_Wr, c3_Wm, c3_b, 256, buf0, buf1);
    launch_gemm<256, 6400>(s, t_P23, buf1, nullptr, buf2, 128, B_, 0, 1638400, 819200, 0);
    // fce1/fce2 (1x1 convs with ELU)
    launch_gemm<64, 128>(s, buf2, fce1_W, fce1_b, buf0, 51200, 1, 0, 0, 0, 1);
    launch_gemm<128, 64>(s, buf0, fce2_W, fce2_b, buf1, 51200, 1, 0, 0, 0, 1);
    // reverse GRU (512 independent rows)
    rev_gru_kernel<<<512, 64, 0, s>>>(buf1, gru_W, gru_U, gru_b, hgru);
    // mu / logvar heads (tanh)
    launch_gemm<64, 64>(s, hgru, mu_W, mu_b, out_mu, 512, 1, 0, 0, 0, 2);
    launch_gemm<64, 64>(s, hgru, lv_W, lv_b, out_lv, 512, 1, 0, 0, 0, 2);
    reparam_kernel<<<128, 256, 0, s>>>(out_mu, out_lv, eps, zlat, 32768);

    // =================== latent graph ===================
    zero_kernel<<<256, 256, 0, s>>>(agg, 65536);
    hinv_scatter_kernel<<<2048, 256, 0, s>>>(zlat, hinv_ei, hinv_ea, agg, 2048);
    launch_gemm<64, 64>(s, agg, trans_W, nullptr, xb2, 1024, 1, 0, 0, 0, 0);

    // =================== GDE (RK4 trajectory) ===================
    gde_kernel<<<512, 128, 0, s>>>(xb2, gde_W1, gde_b1, gde_W2, gde_b2, traj);

    // =================== decoder ===================
    launch_gemm<64, 32>(s, traj, fcd3_W, fcd3_b, buf0, 51200, 1, 0, 0, 0, 1);
    launch_gemm<32, 64>(s, buf0, fcd4_W, fcd4_b, buf1, 51200, 1, 0, 0, 0, 1);
    launch_gemm<128, 6400>(s, P43, buf1, nullptr, buf2, 256, B_, 0, 819200, 1638400, 0);
    gconv<64, 32>(s, buf2, bg3_ei, bg3_ea, 2048, d4_Wr, d4_Wm, d4_b, 256, buf0, buf1);
    launch_gemm<256, 3200>(s, P32, buf1, nullptr, buf2, 512, B_, 0, 819200, 1638400, 0);
    gconv<32, 16>(s, buf2, bg2_ei, bg2_ea, 4096, d3_Wr, d3_Wm, d3_b, 512, buf0, buf1);
    launch_gemm<512, 1600>(s, P21, buf1, nullptr, buf2, 1024, B_, 0, 819200, 1638400, 0);
    gconv<16, 8>(s, buf2, bg1_ei, bg1_ea, 8192, d2_Wr, d2_Wm, d2_b, 1024, buf0, buf1);
    launch_gemm<1024, 800>(s, P10, buf1, nullptr, buf2, 2048, B_, 0, 819200, 1638400, 0);
    // final gconv (8->1) writes phi_h directly into d_out
    gconv<8, 1>(s, buf2, bg_ei, bg_ea, 16384, d1_Wr, d1_Wm, d1_b, 2048, buf0, out_phi_h);

    // =================== output projections ===================
    // l_h[b]   = h_L   (2048x2048) @ ph[b](2048x100)
    launch_gemm<2048, 100>(s, h_L,   out_phi_h, nullptr, out_l_h,   2048, B_, 0, 204800, 204800, 0);
    // phi_t_[b]= H_mat (1024x2048) @ ph[b](2048x100)
    launch_gemm<2048, 100>(s, H_mat, out_phi_h, nullptr, out_phi_t, 1024, B_, 0, 204800, 102400, 0);
}
